// Int4WeightLinear_20650202759472
// MI455X (gfx1250) — compile-verified, matrix operand order checked by reading, not run on hardware
//
#include <hip/hip_runtime.h>

// ---------------- problem constants (from reference setup_inputs) ------------
#define B_DIM   4
#define S_DIM   2048
#define M_TOT   (B_DIM * S_DIM)   // 8192 tokens
#define IN_DIM  4096              // K
#define OUT_DIM 11008             // N
#define GSIZE   32                // quant group size along K
#define NGROUP  (IN_DIM / GSIZE)  // 128 groups

// ---------------- tiling ------------------------------------------------------
#define MT 128                    // CTA M tile
#define NT 128                    // CTA N tile
#define PITCH 40                  // LDS row pitch in halfs (80B) -> conflict-free b128

typedef _Float16 v16h __attribute__((ext_vector_type(16)));
typedef _Float16 v8h  __attribute__((ext_vector_type(8)));
typedef float    v8f  __attribute__((ext_vector_type(8)));
typedef float    f4v  __attribute__((ext_vector_type(4)));
typedef int      i4v  __attribute__((ext_vector_type(4)));

typedef unsigned int u32x4 __attribute__((ext_vector_type(4)));
typedef int          i32x4 __attribute__((ext_vector_type(4)));
typedef int          i32x8 __attribute__((ext_vector_type(8)));

// =============================================================================
// Kernel 1: per-token dynamic symmetric int8 quantization of the activation.
// Stores q_act as f16 (integers in [-128,127] -> exact) so the GEMM's A tile is
// a raw 2D copy that the Tensor Data Mover can DMA into LDS, plus a_scale fp32.
// =============================================================================
__global__ __launch_bounds__(256)
void quant_act_kernel(const float* __restrict__ x,
                      _Float16* __restrict__ qact,
                      float* __restrict__ ascale) {
    __shared__ float red[256];
    const int row = blockIdx.x;
    const int t   = threadIdx.x;

    const f4v* xp = (const f4v*)(x + (size_t)row * IN_DIM + t * 16);
    f4v v[4];
#pragma unroll
    for (int i = 0; i < 4; ++i) v[i] = xp[i];

    float m = 0.f;
#pragma unroll
    for (int i = 0; i < 4; ++i) {
        m = fmaxf(m, fabsf(v[i].x));
        m = fmaxf(m, fabsf(v[i].y));
        m = fmaxf(m, fabsf(v[i].z));
        m = fmaxf(m, fabsf(v[i].w));
    }
    red[t] = m;
    __syncthreads();
#pragma unroll
    for (int s = 128; s > 0; s >>= 1) {
        if (t < s) red[t] = fmaxf(red[t], red[t + s]);
        __syncthreads();
    }
    const float amax = fmaxf(red[0], 1e-12f);
    const float sc   = amax / 127.0f;      // a_scale
    const float inv  = 127.0f / amax;      // 1 / a_scale

    v8h q[2];
#pragma unroll
    for (int i = 0; i < 4; ++i) {
        float e[4] = {v[i].x, v[i].y, v[i].z, v[i].w};
#pragma unroll
        for (int j = 0; j < 4; ++j) {
            float r = rintf(e[j] * inv);               // round-half-even, like jnp.round
            r = fminf(fmaxf(r, -128.f), 127.f);        // clip to int8 range
            q[i >> 1][(i & 1) * 4 + j] = (_Float16)r;  // exact integer in f16
        }
    }
    v8h* dst = (v8h*)(qact + (size_t)row * IN_DIM + t * 16);
    dst[0] = q[0];
    dst[1] = q[1];
    if (t == 0) ascale[row] = sc;
}

// =============================================================================
// TDM: DMA one 128-row x 32-half A tile (f16) from global into LDS, padded by
// hardware to the 40-half (80B) bank-conflict-free pitch:
//   pad_interval = 3 (pad every 16 DWORDs = one 32-half row)
//   pad_amount   = 3 (insert 4 DWORDs = 8 halfs)
// D# built per CDNA5 ISA ch.8 (group0: count/lds/global/type=2; group1: dims).
// =============================================================================
__device__ __forceinline__ void tdm_load_A_tile(const _Float16* gsrc, unsigned lds_off) {
    const unsigned long long ga = (unsigned long long)(uintptr_t)gsrc;
    u32x4 g0;
    g0[0] = 1u;                                                 // count=1, load, no gather
    g0[1] = lds_off;                                            // lds_addr (bytes)
    g0[2] = (unsigned)(ga & 0xffffffffu);                       // global_addr[31:0]
    g0[3] = (unsigned)((ga >> 32) & 0x1ffffffu) | (2u << 30);   // global_addr[56:32], type=2
    i32x8 g1;
    g1[0] = (1 << 16) | (1 << 20) | (3 << 22) | (3 << 25);      // data_size=2B, pad_en, int=16DW, amt=4DW
    g1[1] = (int)((IN_DIM & 0xffff) << 16);                     // tensor_dim0[15:0]
    g1[2] = (int)(((unsigned)IN_DIM >> 16) | ((M_TOT & 0xffff) << 16));   // dim0 hi | dim1 lo
    g1[3] = (int)(((unsigned)M_TOT >> 16) | (32u << 16));       // dim1 hi | tile_dim0=32
    g1[4] = (int)128u;                                          // tile_dim1=128, tile_dim2=0
    g1[5] = (int)IN_DIM;                                        // tensor_dim0_stride[31:0]
    g1[6] = 0;                                                  // stride0 hi, dim1_stride lo
    g1[7] = 0;
    i32x4 z4 = (i32x4){};
#if defined(__clang_major__) && (__clang_major__ >= 23)
    i32x8 z8 = (i32x8){};
    __builtin_amdgcn_tensor_load_to_lds(g0, g1, z4, z4, z8, 0);
#else
    __builtin_amdgcn_tensor_load_to_lds(g0, g1, z4, z4, 0);
#endif
}

// =============================================================================
// Kernel 2: W4(group-dequant) x A8 GEMM via V_WMMA_F32_16X16X32_F16.
// K=32 per WMMA == group size -> exact per-group math:
//   acc += s_g[n] * ( q_a(f16) x (q_w - z_g)(f16) ),  epilogue * a_scale[m].
// Double-buffered LDS: TDM streams the A tile for group g+1 while the waves run
// the 8 WMMAs of group g; weight tile g+1 is register-prefetched before compute
// and converted/stored after. One barrier per K-group.
// =============================================================================
__global__ __launch_bounds__(256)
void w4a8_gemm_kernel(const _Float16* __restrict__ qact,
                      const float* __restrict__ ascale,
                      const int*  __restrict__ weight,   // uint4 values, one per int32
                      const float* __restrict__ scales,  // [OUT][NGROUP]
                      const float* __restrict__ zeros,   // [OUT][NGROUP]
                      float* __restrict__ out) {
    __shared__ __align__(16) _Float16 sA[2][MT * PITCH];  // 2 x 10 KB
    __shared__ __align__(16) _Float16 sB[2][NT * PITCH];  // 2 x 10 KB
    __shared__ float sS[2][NT];                           // group scales, double buffered

    const int t    = threadIdx.x;
    const int n0   = blockIdx.x * NT;
    const int m0   = blockIdx.y * MT;
    const int lane = t & 31;
    const int wave = t >> 5;
    const int wm   = wave & 3;          // M quadrant: wm*32
    const int wn   = wave >> 2;         // N half:     wn*64
    const int lrow = lane & 15;
    const int hb   = lane >> 4;         // ISA 16-bit layout: hi lanes hold K+8 / M+8

    v8f acc[2][4];
#pragma unroll
    for (int i = 0; i < 2; ++i)
#pragma unroll
        for (int j = 0; j < 4; ++j) acc[i][j] = (v8f){};

    const int srow = t >> 1;            // staging row 0..127
    const int sk   = (t & 1) * 16;      // staging k-offset within the 32-wide tile
    const int*   wrow = weight + (size_t)(n0 + srow) * IN_DIM + sk;
    const float* zrow = zeros  + (size_t)(n0 + srow) * NGROUP;

    // ---- staging helpers (double-buffered, register-prefetched weights) -----
    i4v   wreg[4];
    float zreg = 0.f, sreg = 0.f;

    auto issue_stage = [&](int g) {     // start loads for group g (no use yet)
        if (wave == 0)
            tdm_load_A_tile(qact + (size_t)m0 * IN_DIM + g * GSIZE,
                            (unsigned)(uintptr_t)&sA[g & 1][0]);
        const i4v* wp = (const i4v*)(wrow + g * GSIZE);
#pragma unroll
        for (int i = 0; i < 4; ++i) wreg[i] = wp[i];
        zreg = zrow[g];
        if (t < NT) sreg = scales[(size_t)(n0 + t) * NGROUP + g];
    };
    auto finish_stage = [&](int g) {    // dequant + publish into LDS buffer g&1
        v8h lo, hi;
#pragma unroll
        for (int i = 0; i < 8; ++i) {
            lo[i] = (_Float16)((float)wreg[i >> 2][i & 3] - zreg);        // exact ints
            hi[i] = (_Float16)((float)wreg[2 + (i >> 2)][i & 3] - zreg);
        }
        v8h* dst = (v8h*)(&sB[g & 1][0] + srow * PITCH + sk);
        dst[0] = lo;
        dst[1] = hi;
        if (t < NT) sS[g & 1][t] = sreg;
        if (wave == 0) __builtin_amdgcn_s_wait_tensorcnt(0);  // A tile g landed
    };

    // ---- prologue: stage group 0 --------------------------------------------
    issue_stage(0);
    finish_stage(0);
    __syncthreads();

    for (int g = 0; g < NGROUP; ++g) {
        const int buf = g & 1;
        if (g + 1 < NGROUP) issue_stage(g + 1);   // overlap with compute below

        // ---- A fragments: v16h = two contiguous 16B K-runs per ISA layout
        v16h af[2];
#pragma unroll
        for (int mf = 0; mf < 2; ++mf) {
            const _Float16* pa = &sA[buf][0] + (wm * 32 + mf * 16 + lrow) * PITCH;
            v8h c0 = *(const v8h*)(pa + 8 * hb);        // K =  8*hb .. 8*hb+7
            v8h c1 = *(const v8h*)(pa + 16 + 8 * hb);   // K = 16+8*hb .. 23+8*hb
            af[mf] = __builtin_shufflevector(c0, c1, 0, 1, 2, 3, 4, 5, 6, 7,
                                                     8, 9, 10, 11, 12, 13, 14, 15);
        }
        // ---- 4 N-fragments x 2 M-fragments of WMMA + per-group scale FMA
#pragma unroll
        for (int nf = 0; nf < 4; ++nf) {
            const _Float16* pb = &sB[buf][0] + (wn * 64 + nf * 16 + lrow) * PITCH;
            v8h b0 = *(const v8h*)(pb + 8 * hb);
            v8h b1 = *(const v8h*)(pb + 16 + 8 * hb);
            v16h bf = __builtin_shufflevector(b0, b1, 0, 1, 2, 3, 4, 5, 6, 7,
                                                      8, 9, 10, 11, 12, 13, 14, 15);
            const float s = sS[buf][wn * 64 + nf * 16 + lrow];  // D tile: lane -> N col
#pragma unroll
            for (int mf = 0; mf < 2; ++mf) {
                v8f zero = (v8f){};
                v8f d = __builtin_amdgcn_wmma_f32_16x16x32_f16(
                    false, af[mf], false, bf, (short)0, zero, false, false);
                acc[mf][nf] += d * s;                   // hides under XDL co-exec
            }
        }
        if (g + 1 < NGROUP) finish_stage(g + 1);        // waits loads after compute
        __syncthreads();
    }

    // ---- epilogue: out[m][n] = a_scale[m] * acc
#pragma unroll
    for (int mf = 0; mf < 2; ++mf) {
#pragma unroll
        for (int nf = 0; nf < 4; ++nf) {
            const int col = n0 + wn * 64 + nf * 16 + lrow;
#pragma unroll
            for (int r = 0; r < 8; ++r) {
                const int row = m0 + wm * 32 + mf * 16 + hb * 8 + r;  // D: M = r + 8*hb
                out[(size_t)row * OUT_DIM + col] = acc[mf][nf][r] * ascale[row];
            }
        }
    }
}

// =============================================================================
extern "C" void kernel_launch(void* const* d_in, const int* in_sizes, int n_in,
                              void* d_out, int out_size, void* d_ws, size_t ws_size,
                              hipStream_t stream) {
    const float* x      = (const float*)d_in[0];
    const int*   weight = (const int*)d_in[1];    // int input -> const int*
    const float* scales = (const float*)d_in[2];
    const float* zeros  = (const float*)d_in[3];
    float*       out    = (float*)d_out;

    _Float16* qact   = (_Float16*)d_ws;                                        // 64 MB
    float*    ascale = (float*)((char*)d_ws + (size_t)M_TOT * IN_DIM * 2);     // +32 KB

    quant_act_kernel<<<M_TOT, 256, 0, stream>>>(x, qact, ascale);

    dim3 grid(OUT_DIM / NT, M_TOT / MT);   // 86 x 64 CTAs
    w4a8_gemm_kernel<<<grid, 256, 0, stream>>>(qact, ascale, weight, scales, zeros, out);
}
